// SparseAutoEncoder_43319040147806
// MI455X (gfx1250) — matrix operand order repaired and checked by reading, not compiled, not run on hardware
//
#include <hip/hip_runtime.h>

// ---------------------------------------------------------------------------
// SparseAutoEncoder fused kernels for gfx1250 (MI455X).
//   K0: convert We f32 -> bf16 (Webf, 33 MB in d_ws)   [one-time, HBM-rate]
//   K1: transpose Wd [1024,16384] -> WdT f32 (67 MB in d_ws)
//   K2: per 64-row tile: encode GEMM via v_wmma_f32_16x16x32_bf16 with the
//       x-tile LDS-resident (bf16) and 64-col Webf chunks staged with
//       GLOBAL_LOAD_ASYNC_TO_LDS_B128 (ASYNCcnt), double-purposed behind the
//       fused streaming top-32 epilogue; then sparse decode from WdT.
// ---------------------------------------------------------------------------

typedef __attribute__((ext_vector_type(16))) __bf16 v16bf;
typedef __attribute__((ext_vector_type(8)))  __bf16 v8bf;
typedef __attribute__((ext_vector_type(8)))  float  v8f;

constexpr int kB    = 8192;
constexpr int kNin  = 1024;
constexpr int kNhid = 16384;
constexpr int kNout = 1024;
constexpr int kTopK = 32;

constexpr int BM = 64;            // rows per workgroup
constexpr int BN = 64;            // cols per N-chunk
constexpr int NCHUNK = kNhid / BN;
constexpr int NTHREADS = 256;     // 8 wave32s: 4 (M) x 2 (N)
constexpr int XS_STRIDE = kNin + 8;   // bf16 halves; 2064B row -> 4-bank shift
constexpr int HS_STRIDE = BN + 1;     // floats; breaks 2-way stage conflicts

// LDS carve (bytes):
constexpr size_t OFF_XS   = 0;                                    // 132096
constexpr size_t OFF_BS   = OFF_XS + (size_t)BM * XS_STRIDE * 2;  // 132096
constexpr size_t OFF_HS   = OFF_BS + (size_t)BN * XS_STRIDE * 2;  // 16640
constexpr size_t OFF_TV   = OFF_HS + (size_t)BM * HS_STRIDE * 4;  // 8192
constexpr size_t OFF_TI   = OFF_TV + (size_t)BM * kTopK * 4;      // 8192
constexpr size_t OFF_MASK = OFF_TI + (size_t)BM * kTopK * 4;      // 512
constexpr size_t SMEM_BYTES = OFF_MASK + (size_t)BM * 2 * 4;      // 297728 <= 320K

// workspace carve (bytes): WdT f32 then Webf bf16
constexpr size_t WS_OFF_WDT  = 0;                                  // 67108864
constexpr size_t WS_OFF_WEBF = (size_t)kNhid * kNout * 4;          // +33554432

__device__ __forceinline__ void store4bf(__bf16* p, float4 v) {
  union { __bf16 h[4]; unsigned long long u; } pk;
  pk.h[0] = (__bf16)v.x; pk.h[1] = (__bf16)v.y;
  pk.h[2] = (__bf16)v.z; pk.h[3] = (__bf16)v.w;
  *(unsigned long long*)p = pk.u;
}

__device__ __forceinline__ v16bf cat8(v8bf lo, v8bf hi) {
  return __builtin_shufflevector(lo, hi, 0,1,2,3,4,5,6,7,8,9,10,11,12,13,14,15);
}

__device__ __forceinline__ void async_wait0() {
  asm volatile("s_wait_asynccnt 0x0" ::: "memory");
}

// Issue async global->LDS copy of one 64x1024 bf16 We chunk into bs
// (per-lane 16B packets; LDS offset = low 32 bits of generic shared address).
__device__ __forceinline__ void stage_we_async(const __bf16* __restrict__ Webf,
                                               __bf16* bs, int nbase, int tid) {
#pragma unroll
  for (int i = 0; i < (BN * kNin) / (8 * NTHREADS); ++i) {   // 32 packets/thread
    const int g = tid + i * NTHREADS;
    const int n = g >> 7;              // 128 x 16B packets per row
    const int k = (g & 127) * 8;       // half index, multiple of 8
    const unsigned long long src =
        (unsigned long long)(size_t)(Webf + (size_t)(nbase + n) * kNin + k);
    const unsigned dst = (unsigned)(size_t)(bs + (size_t)n * XS_STRIDE + k);
    asm volatile("global_load_async_to_lds_b128 %0, %1, off"
                 :: "v"(dst), "v"(src) : "memory");
  }
}

// ---------------------------------------------------------------------------
// Kernel 0: We f32 -> bf16
// ---------------------------------------------------------------------------
__global__ void sae_we_to_bf16(const float* __restrict__ We,
                               __bf16* __restrict__ Webf) {
  const size_t g = ((size_t)blockIdx.x * blockDim.x + threadIdx.x) * 4;
  float4 v = *(const float4*)(We + g);
  store4bf(Webf + g, v);
}

// ---------------------------------------------------------------------------
// Kernel 1: Wd [NOUT, NHID] -> WdT [NHID, NOUT]
// ---------------------------------------------------------------------------
__global__ void sae_wd_transpose(const float* __restrict__ Wd,
                                 float* __restrict__ WdT) {
  __shared__ float tile[32][33];
  const int j0 = blockIdx.x * 32;   // NHID
  const int o0 = blockIdx.y * 32;   // NOUT
  const int tx = threadIdx.x, ty = threadIdx.y;   // (32, 8)
#pragma unroll
  for (int i = 0; i < 4; ++i)
    tile[ty + 8 * i][tx] = Wd[(size_t)(o0 + ty + 8 * i) * kNhid + j0 + tx];
  __syncthreads();
#pragma unroll
  for (int i = 0; i < 4; ++i)
    WdT[(size_t)(j0 + ty + 8 * i) * kNout + o0 + tx] = tile[tx][ty + 8 * i];
}

// ---------------------------------------------------------------------------
// Kernel 2: fused encode-GEMM + streaming top-32 + sparse decode
// ---------------------------------------------------------------------------
__global__ void __launch_bounds__(NTHREADS)
sae_fused(const float* __restrict__ x,  const __bf16* __restrict__ Webf,
          const float* __restrict__ be, const float* __restrict__ WdT,
          const float* __restrict__ bd, float* __restrict__ out) {
  extern __shared__ char smem[];
  __bf16*       xs     = (__bf16*)(smem + OFF_XS);
  __bf16*       bs     = (__bf16*)(smem + OFF_BS);
  float*        hstage = (float*) (smem + OFF_HS);
  float*        topVal = (float*) (smem + OFF_TV);
  int*          topIdx = (int*)   (smem + OFF_TI);
  unsigned int* mask   = (unsigned int*)(smem + OFF_MASK);

  const int tid  = threadIdx.x;
  const int lane = tid & 31;
  const int wave = tid >> 5;
  const int wm   = wave & 3;     // 0..3 -> rows wm*16
  const int wn   = wave >> 2;    // 0..1 -> cols wn*32
  const int nl   = lane & 15;
  const int hi   = lane >> 4;    // lane half-group
  const int rowbase = blockIdx.x * BM;

  // kick off async staging of chunk 0 while we build the x tile
  stage_we_async(Webf, bs, 0, tid);

  // ---- load x tile once: BM x NIN f32 -> bf16 LDS ----
#pragma unroll 4
  for (int i = 0; i < (BM * kNin) / (4 * NTHREADS); ++i) {   // 64 iters
    const int g = tid + i * NTHREADS;
    const int m = (g * 4) >> 10, k = (g * 4) & (kNin - 1);
    float4 v = *(const float4*)(x + (size_t)(rowbase + m) * kNin + k);
    store4bf(xs + (size_t)m * XS_STRIDE + k, v);
  }
  for (int i = tid; i < BM * kTopK; i += NTHREADS) {
    topVal[i] = -3.0e38f; topIdx[i] = 0;
  }
  async_wait0();
  __syncthreads();

  // ---- stream N chunks ----
  for (int nc = 0; nc < NCHUNK; ++nc) {
    const int nbase = nc * BN;

    v8f c0 = {0.f,0.f,0.f,0.f,0.f,0.f,0.f,0.f};
    v8f c1 = {0.f,0.f,0.f,0.f,0.f,0.f,0.f,0.f};

    const __bf16* arow = xs + (size_t)(wm * 16 + nl) * XS_STRIDE + hi * 16;
    const __bf16* brow = bs + (size_t)(wn * 32 + nl) * XS_STRIDE + hi * 16;

#pragma unroll 2
    for (int k0 = 0; k0 < kNin; k0 += 32) {
      // A frag: K contiguous per half-group {hi*16..+15}
      v16bf a = cat8(*(const v8bf*)(arow + k0),
                     *(const v8bf*)(arow + k0 + 8));
      // B frags (same k order as A): N = lane, K contiguous per half-group
      v16bf b0 = cat8(*(const v8bf*)(brow + k0),
                      *(const v8bf*)(brow + k0 + 8));
      v16bf b1 = cat8(*(const v8bf*)(brow + 16 * XS_STRIDE + k0),
                      *(const v8bf*)(brow + 16 * XS_STRIDE + k0 + 8));
      c0 = __builtin_amdgcn_wmma_f32_16x16x32_bf16(false, a, false, b0,
                                                   (short)0, c0, false, false);
      c1 = __builtin_amdgcn_wmma_f32_16x16x32_bf16(false, a, false, b1,
                                                   (short)0, c1, false, false);
    }

    // epilogue: + be, stage tile (C layout: VGPR g -> M=g / g+8, col = lane&15)
#pragma unroll
    for (int g = 0; g < 8; ++g) {
      const int m = wm * 16 + g + hi * 8;
      const int n = wn * 32 + nl;
      hstage[m * HS_STRIDE + n]      = c0[g] + be[nbase + n];
      hstage[m * HS_STRIDE + n + 16] = c1[g] + be[nbase + n + 16];
    }
    if (tid < BM * 2) mask[tid] = 0u;   // prev chunk's masks already consumed
    __syncthreads();                    // all waves done reading bs + hstage ready

    // overlap: async-stage next chunk into bs behind the top-k epilogue
    if (nc + 1 < NCHUNK) stage_we_async(Webf, bs, nbase + BN, tid);

    // parallel pre-filter against per-row running min (topVal[m][0])
#pragma unroll 4
    for (int i = 0; i < (BM * BN) / NTHREADS; ++i) {   // 16 elems/thread
      const int e = tid * 16 + i;
      const int m = e >> 6, n = e & 63;
      if (hstage[m * HS_STRIDE + n] > topVal[m * kTopK])
        atomicOr(&mask[m * 2 + (n >> 5)], 1u << (n & 31));
    }
    __syncthreads();

    // per-row owner thread: insert candidates into sorted-ascending top-32
    if (tid < BM) {
      float* tv = topVal + tid * kTopK;
      int*   ti = topIdx + tid * kTopK;
      for (int w = 0; w < 2; ++w) {
        unsigned int bits = mask[tid * 2 + w];
        while (bits) {
          const int b = __builtin_ctz(bits);
          bits &= bits - 1;
          const int n = w * 32 + b;
          const float v = hstage[tid * HS_STRIDE + n];
          if (v > tv[0]) {
            int p = 1;
            while (p < kTopK && tv[p] < v) {
              tv[p - 1] = tv[p]; ti[p - 1] = ti[p]; ++p;
            }
            tv[p - 1] = v; ti[p - 1] = nbase + n;
          }
        }
      }
    }
    async_wait0();        // this wave's staging of chunk nc+1 has landed
    __syncthreads();      // -> every wave's slice landed; bs valid
  }

  // ---- sparse decode: out[m,:] = bd + sum_t topVal * WdT[topIdx,:] ----
  const int o = tid * 4;
  for (int m = 0; m < BM; ++m) {
    const float4 b4 = *(const float4*)(bd + o);
    float ax = b4.x, ay = b4.y, az = b4.z, aw = b4.w;
#pragma unroll 4
    for (int t = 0; t < kTopK; ++t) {
      const float v = topVal[m * kTopK + t];
      const int   j = topIdx[m * kTopK + t];
      const float4 w = *(const float4*)(WdT + (size_t)j * kNout + o);
      ax += v * w.x; ay += v * w.y; az += v * w.z; aw += v * w.w;
    }
    float4 r; r.x = ax; r.y = ay; r.z = az; r.w = aw;
    *(float4*)(out + (size_t)(rowbase + m) * kNout + o) = r;
  }
}

// ---------------------------------------------------------------------------
extern "C" void kernel_launch(void* const* d_in, const int* in_sizes, int n_in,
                              void* d_out, int out_size, void* d_ws, size_t ws_size,
                              hipStream_t stream) {
  (void)in_sizes; (void)n_in; (void)out_size; (void)ws_size;
  const float* x  = (const float*)d_in[0];
  const float* We = (const float*)d_in[1];
  const float* be = (const float*)d_in[2];
  const float* Wd = (const float*)d_in[3];
  const float* bd = (const float*)d_in[4];
  float* out = (float*)d_out;

  float*  WdT  = (float*) ((char*)d_ws + WS_OFF_WDT);   // 67 MB
  __bf16* Webf = (__bf16*)((char*)d_ws + WS_OFF_WEBF);  // 33 MB

  hipFuncSetAttribute(reinterpret_cast<const void*>(sae_fused),
                      hipFuncAttributeMaxDynamicSharedMemorySize,
                      (int)SMEM_BYTES);

  sae_we_to_bf16<<<dim3((kNhid * kNin) / (4 * NTHREADS)), dim3(NTHREADS), 0,
                   stream>>>(We, Webf);
  sae_wd_transpose<<<dim3(kNhid / 32, kNout / 32), dim3(32, 8), 0, stream>>>(Wd, WdT);
  sae_fused<<<dim3(kB / BM), dim3(NTHREADS), SMEM_BYTES, stream>>>(
      x, Webf, be, WdT, bd, out);
}
